// MultiHeadAttention_79534204387726
// MI455X (gfx1250) — matrix operand order repaired and checked by reading, not compiled, not run on hardware
//
#include <hip/hip_runtime.h>
#include <hip/hip_bf16.h>

typedef __attribute__((ext_vector_type(16))) _Float16 v16h;
typedef __attribute__((ext_vector_type(8)))  float    v8f;
typedef __attribute__((ext_vector_type(4)))  unsigned tdm_g0_t;
typedef __attribute__((ext_vector_type(8)))  unsigned tdm_g1_t;

namespace {
constexpr int Bc = 4, Sc = 1024, Ec = 1024, Hc = 16, DHc = 64;
}

__device__ __forceinline__ v8f zero8() {
  v8f z;
#pragma unroll
  for (int i = 0; i < 8; ++i) z[i] = 0.0f;
  return z;
}

__device__ __forceinline__ v8f wmma_f16(v16h a, v16h b, v8f c) {
  // (neg_a, A, neg_b, B, c_mod, C, reuse_a, reuse_b)
  return __builtin_amdgcn_wmma_f32_16x16x32_f16(false, a, false, b, (short)0, c,
                                                false, false);
}

// A-matrix 16x32 f16 fragment: lane m = L&15, kb = (L>=16)?16:0,
// halves j -> K = kb+j.
__device__ __forceinline__ v16h load_a_frag(const _Float16* rowp, int kb) {
  v16h a;
#pragma unroll
  for (int j = 0; j < 16; ++j) a[j] = rowp[kb + j];
  return a;
}

// B-matrix 32x16 f16 fragment: lane n = L&15, kb = (L>=16)?16:0, halves j -> K=kb+j.
__device__ __forceinline__ v16h load_b_frag(const _Float16* colp, int kb) {
  v16h b;
#pragma unroll
  for (int j = 0; j < 16; ++j) b[j] = colp[kb + j];
  return b;
}

// ---------------------------------------------------------------------------
// Tensor Data Mover: 2D f16 tile (tile_d0 elems/row x tile_d1 rows, global row
// stride = stride_d0 elems) -> LDS at lds_off, inserting 1 DWORD of padding
// every 2^(pad_code+1) DWORDs stored (so LDS row pitch = row bytes + 4).
// Descriptor layout per CDNA5 ISA.
// Issue from ONE wave; completion via s_wait_tensorcnt + workgroup barrier.
// ---------------------------------------------------------------------------
__device__ __forceinline__ void tdm_load_2d_f16(unsigned lds_off, const void* gptr,
                                                unsigned tile_d0, unsigned tile_d1,
                                                unsigned stride_d0,
                                                unsigned pad_code) {
  const unsigned long long ga = (unsigned long long)gptr;
  tdm_g0_t g0;
  g0[0] = 1u;                                             // count=1 (user D#)
  g0[1] = lds_off;                                        // lds_addr
  g0[2] = (unsigned)(ga & 0xffffffffu);                   // global_addr[31:0]
  g0[3] = (unsigned)((ga >> 32) & 0x1ffffffu) | (2u << 30); // addr[56:32]|type=2
  tdm_g1_t g1;
  g1[0] = (1u << 16)            // data_size = 1 -> 2-byte elements
        | (1u << 20)            // pad_enable
        | (pad_code << 22);     // pad_interval; pad_amount=0 -> 1 DWORD
  g1[1] = (tile_d0 & 0xffffu) << 16;                      // tensor_dim0 = tile_d0
  g1[2] = (tile_d0 >> 16) | ((tile_d1 & 0xffffu) << 16);  // tensor_dim1 = tile_d1
  g1[3] = (tile_d1 >> 16) | (tile_d0 << 16);              // tile_dim0
  g1[4] = tile_d1;                                        // tile_dim1; tile_dim2=0
  g1[5] = stride_d0;                                      // tensor_dim0_stride lo
  g1[6] = 0;                                              // stride hi / dim1_stride
  g1[7] = 0;
  asm volatile("tensor_load_to_lds %0, %1" : : "s"(g0), "s"(g1) : "memory");
}

// ---------------------------------------------------------------------------
// Kernel 1: QKV projection.  Y = X @ Wq^T + bq for X in {q,k,v} (M = 3*B*S),
// f32 in -> f16 out in head-major [b][h][s][d] layout; Q pre-scaled by 1/8.
// (Tiles staged through VGPRs: the f32->f16 convert must ride the load path.)
// ---------------------------------------------------------------------------
__global__ __launch_bounds__(256)
void qkv_proj_kernel(const float* __restrict__ q, const float* __restrict__ k,
                     const float* __restrict__ v, const float* __restrict__ W,
                     const float* __restrict__ bias,
                     _Float16* __restrict__ oq, _Float16* __restrict__ ok,
                     _Float16* __restrict__ ov) {
  __shared__ _Float16 As[64][40];    // [m][k] tile (f16-converted)
  __shared__ _Float16 Bt[128][40];   // [n][k] tile: W rows are already K-major

  const int t = threadIdx.x;
  const int lane = t & 31, wid = t >> 5;
  const int wm = (wid >> 2) * 32;
  const int wn = (wid & 3) * 32;
  const int block_m = blockIdx.x * 64;
  const int block_n = blockIdx.y * 128;

  const int tensor = block_m >> 12;  // which of q/k/v (4096 rows each)
  const int m_in = block_m & 4095;
  const float* X; _Float16* O; float scl;
  if (tensor == 0)      { X = q; O = oq; scl = 0.125f; }
  else if (tensor == 1) { X = k; O = ok; scl = 1.0f;   }
  else                  { X = v; O = ov; scl = 1.0f;   }

  v8f acc[2][2];
#pragma unroll
  for (int i = 0; i < 2; ++i)
#pragma unroll
    for (int j = 0; j < 2; ++j) acc[i][j] = zero8();

  const int ar = t >> 2, ac = (t & 3) * 8;    // A tile: 64x32, 8 elems/thread
  const int br = t >> 1, bc = (t & 1) * 16;   // B tile: 128x32, 16 elems/thread
  const int l16 = lane & 15, hi = lane >> 4;
  const int kbA = hi * 16, kbB = hi * 16;

  for (int k0 = 0; k0 < Ec; k0 += 32) {
    __syncthreads();
#pragma unroll
    for (int j = 0; j < 8; ++j)
      As[ar][ac + j] = (_Float16)X[(size_t)(m_in + ar) * Ec + k0 + ac + j];
#pragma unroll
    for (int j = 0; j < 16; ++j)
      Bt[br][bc + j] = (_Float16)W[(size_t)(block_n + br) * Ec + k0 + bc + j];
    __syncthreads();

    v16h a0 = load_a_frag(&As[wm + l16][0], kbA);
    v16h a1 = load_a_frag(&As[wm + 16 + l16][0], kbA);
    v16h b0 = load_b_frag(&Bt[wn + l16][0], kbB);
    v16h b1 = load_b_frag(&Bt[wn + 16 + l16][0], kbB);
    acc[0][0] = wmma_f16(a0, b0, acc[0][0]);
    acc[0][1] = wmma_f16(a0, b1, acc[0][1]);
    acc[1][0] = wmma_f16(a1, b0, acc[1][0]);
    acc[1][1] = wmma_f16(a1, b1, acc[1][1]);
  }

  // Epilogue: bias, scale, scatter to [b][h][s][d] f16.
#pragma unroll
  for (int i = 0; i < 2; ++i)
#pragma unroll
    for (int j = 0; j < 2; ++j) {
      const int n_g = block_n + wn + j * 16 + l16;
      const float bval = bias[n_g];
      const int h = n_g >> 6, d = n_g & 63;
#pragma unroll
      for (int r = 0; r < 8; ++r) {
        const int m_g = m_in + wm + i * 16 + r + hi * 8;  // row in [0,4096)
        const int bb = m_g >> 10, ss = m_g & 1023;
        const float val = (acc[i][j][r] + bval) * scl;
        O[(((size_t)(bb * Hc + h)) * Sc + ss) * DHc + d] = (_Float16)val;
      }
    }
}

// ---------------------------------------------------------------------------
// Kernel 2: flash attention. One block = 128 q rows of one (b,h); 8 waves each
// own 16 q rows; stream K/V in 32-key tiles. K tile arrives via the Tensor
// Data Mover (pad 1 DWORD per 64-half row -> pitch 66); V is transposed
// manually (TDM cannot transpose).
// ---------------------------------------------------------------------------
__global__ __launch_bounds__(256)
void flash_attn_kernel(const _Float16* __restrict__ Q,
                       const _Float16* __restrict__ K,
                       const _Float16* __restrict__ V,
                       _Float16* __restrict__ Oa) {
  __shared__ _Float16 Ks[32][66];        // [key][d], TDM-padded pitch
  __shared__ _Float16 Vt[64][40];        // [d][key] (transposed)
  __shared__ _Float16 Ps[8][16][32];     // per-wave P scratch (C->A relayout)

  const int t = threadIdx.x, lane = t & 31, wid = t >> 5;
  const int l16 = lane & 15, hi = lane >> 4;
  const int kbA = hi * 16, kbB = hi * 16;
  const int bh = blockIdx.y;             // b*H + h
  const int b = bh >> 4, h = bh & 15;
  const int q0 = blockIdx.x * 128;

  const size_t base = (size_t)bh * Sc * DHc;
  const _Float16* Qp = Q + base;
  const _Float16* Kp = K + base;
  const _Float16* Vp = V + base;
  const unsigned ks_lds = (unsigned)(uintptr_t)(&Ks[0][0]);

  // Q fragments held in registers for the whole pass (Q already holds 1/8 scale)
  const int qrow = q0 + wid * 16 + l16;
  v16h qa0 = load_a_frag(Qp + (size_t)qrow * DHc, kbA);        // d = 0..31
  v16h qa1 = load_a_frag(Qp + (size_t)qrow * DHc + 32, kbA);   // d = 32..63

  v8f acc[4];
#pragma unroll
  for (int i = 0; i < 4; ++i) acc[i] = zero8();
  float mrow[8], lrow[8];
#pragma unroll
  for (int r = 0; r < 8; ++r) { mrow[r] = -3.0e38f; lrow[r] = 0.0f; }

  const int krow = t >> 3, kcol = (t & 7) * 8;  // 32x64 tile, 8 halves/thread

  for (int kk = 0; kk < Sc; kk += 32) {
    __syncthreads();
    if (wid == 0) {
      // K tile: 32 rows x 64 f16, global row stride 64, pad every 32 DWORDs.
      tdm_load_2d_f16(ks_lds, Kp + (size_t)kk * DHc, DHc, 32, DHc, 4);
    }
#pragma unroll
    for (int j = 0; j < 8; ++j)
      Vt[kcol + j][krow] = Vp[(size_t)(kk + krow) * DHc + kcol + j];
    if (kk + 32 < Sc)
      __builtin_prefetch(Vp + (size_t)(kk + 32 + krow) * DHc + kcol, 0, 1);
    if (wid == 0) __builtin_amdgcn_s_wait_tensorcnt(0);
    __syncthreads();

    // scores: S[16q x 32k] = Q(16x64) @ K^T(64x32), two 16-col key groups
    v16h kb0l = load_b_frag(&Ks[l16][0],       kbB);
    v16h kb0h = load_b_frag(&Ks[l16][32],      kbB);
    v16h kb1l = load_b_frag(&Ks[16 + l16][0],  kbB);
    v16h kb1h = load_b_frag(&Ks[16 + l16][32], kbB);
    v8f s0 = wmma_f16(qa1, kb0h, wmma_f16(qa0, kb0l, zero8()));
    v8f s1 = wmma_f16(qa1, kb1h, wmma_f16(qa0, kb1l, zero8()));

    // online softmax; each C row lives in one 16-lane half -> width-16 shuffles
#pragma unroll
    for (int r = 0; r < 8; ++r) {
      float mx = fmaxf(s0[r], s1[r]);
#pragma unroll
      for (int off = 8; off >= 1; off >>= 1)
        mx = fmaxf(mx, __shfl_xor(mx, off, 16));
      const float mnew  = fmaxf(mrow[r], mx);
      const float alpha = __expf(mrow[r] - mnew);
      const float p0 = __expf(s0[r] - mnew);
      const float p1 = __expf(s1[r] - mnew);
      float rs = p0 + p1;
#pragma unroll
      for (int off = 8; off >= 1; off >>= 1)
        rs += __shfl_xor(rs, off, 16);
      lrow[r] = lrow[r] * alpha + rs;
      mrow[r] = mnew;
#pragma unroll
      for (int tt = 0; tt < 4; ++tt) acc[tt][r] *= alpha;
      const int pr = r + hi * 8;                 // C layout row
      Ps[wid][pr][l16]      = (_Float16)p0;
      Ps[wid][pr][16 + l16] = (_Float16)p1;
    }

    // O += P(16x32) @ V(32x64); same-wave DS ops are in-order (no barrier)
    v16h pa = load_a_frag(&Ps[wid][l16][0], kbA);
#pragma unroll
    for (int tt = 0; tt < 4; ++tt) {
      v16h vb = load_b_frag(&Vt[tt * 16 + l16][0], kbB);
      acc[tt] = wmma_f16(pa, vb, acc[tt]);
    }
  }

  // normalize and store f16 in [b][s][e] layout for the output projection
#pragma unroll
  for (int r = 0; r < 8; ++r) {
    const float inv = 1.0f / lrow[r];
    const int s = q0 + wid * 16 + r + hi * 8;
#pragma unroll
    for (int tt = 0; tt < 4; ++tt) {
      const int e = h * 64 + tt * 16 + l16;
      Oa[((size_t)(b * Sc + s)) * Ec + e] = (_Float16)(acc[tt][r] * inv);
    }
  }
}

// ---------------------------------------------------------------------------
// Kernel 3: output projection.  out = A(f16) @ Wo^T + bo, f32 out.
// A tile (already f16) arrives via TDM: 64 rows x 32 f16, row stride 1024,
// pad every 16 DWORDs -> LDS pitch 34 halves.
// ---------------------------------------------------------------------------
__global__ __launch_bounds__(256)
void out_proj_kernel(const _Float16* __restrict__ A, const float* __restrict__ W,
                     const float* __restrict__ bias, float* __restrict__ out) {
  __shared__ _Float16 As[64][34];    // TDM-padded pitch
  __shared__ _Float16 Bt[128][40];

  const int t = threadIdx.x, lane = t & 31, wid = t >> 5;
  const int wm = (wid >> 2) * 32, wn = (wid & 3) * 32;
  const int block_m = blockIdx.x * 64, block_n = blockIdx.y * 128;
  const int l16 = lane & 15, hi = lane >> 4;
  const int kbA = hi * 16, kbB = hi * 16;
  const unsigned as_lds = (unsigned)(uintptr_t)(&As[0][0]);

  v8f acc[2][2];
#pragma unroll
  for (int i = 0; i < 2; ++i)
#pragma unroll
    for (int j = 0; j < 2; ++j) acc[i][j] = zero8();

  const int br = t >> 1, bc = (t & 1) * 16;

  for (int k0 = 0; k0 < Ec; k0 += 32) {
    __syncthreads();
    if (wid == 0) {
      // A tile: 64 rows x 32 f16, row stride Ec, pad every 16 DWORDs (code 3).
      tdm_load_2d_f16(as_lds, A + (size_t)block_m * Ec + k0, 32, 64, Ec, 3);
    }
#pragma unroll
    for (int j = 0; j < 16; ++j)
      Bt[br][bc + j] = (_Float16)W[(size_t)(block_n + br) * Ec + k0 + bc + j];
    if (wid == 0) __builtin_amdgcn_s_wait_tensorcnt(0);
    __syncthreads();

    v16h a0 = load_a_frag(&As[wm + l16][0], kbA);
    v16h a1 = load_a_frag(&As[wm + 16 + l16][0], kbA);
    v16h b0 = load_b_frag(&Bt[wn + l16][0], kbB);
    v16h b1 = load_b_frag(&Bt[wn + 16 + l16][0], kbB);
    acc[0][0] = wmma_f16(a0, b0, acc[0][0]);
    acc[0][1] = wmma_f16(a0, b1, acc[0][1]);
    acc[1][0] = wmma_f16(a1, b0, acc[1][0]);
    acc[1][1] = wmma_f16(a1, b1, acc[1][1]);
  }

#pragma unroll
  for (int i = 0; i < 2; ++i)
#pragma unroll
    for (int j = 0; j < 2; ++j) {
      const int n_g = block_n + wn + j * 16 + l16;
      const float bval = bias[n_g];
#pragma unroll
      for (int r = 0; r < 8; ++r) {
        const int m_g = block_m + wm + i * 16 + r + hi * 8;
        out[(size_t)m_g * Ec + n_g] = acc[i][j][r] + bval;
      }
    }
}

extern "C" void kernel_launch(void* const* d_in, const int* in_sizes, int n_in,
                              void* d_out, int out_size, void* d_ws, size_t ws_size,
                              hipStream_t stream) {
  const float* q  = (const float*)d_in[0];
  const float* k  = (const float*)d_in[1];
  const float* v  = (const float*)d_in[2];
  const float* Wq = (const float*)d_in[3];
  const float* bq = (const float*)d_in[4];
  const float* Wo = (const float*)d_in[5];
  const float* bo = (const float*)d_in[6];
  float* out = (float*)d_out;

  const size_t per = (size_t)Bc * Hc * Sc * DHc;   // elems per projected tensor
  _Float16* wsQ = (_Float16*)d_ws;
  _Float16* wsK = wsQ + per;
  _Float16* wsV = wsK + per;
  _Float16* wsA = wsV + per;                       // attn output [B*S][E] f16

  qkv_proj_kernel<<<dim3(3 * Bc * Sc / 64, Ec / 128), dim3(256), 0, stream>>>(
      q, k, v, Wq, bq, wsQ, wsK, wsV);
  flash_attn_kernel<<<dim3(Sc / 128, Bc * Hc), dim3(256), 0, stream>>>(
      wsQ, wsK, wsV, wsA);
  out_proj_kernel<<<dim3(Bc * Sc / 64, Ec / 128), dim3(256), 0, stream>>>(
      wsA, Wo, bo, out);
}
